// TextEncoder_48112223650531
// MI455X (gfx1250) — matrix operand order repaired and checked by reading, not compiled, not run on hardware
//
#include <hip/hip_runtime.h>

typedef __attribute__((ext_vector_type(16))) _Float16 v16h;
typedef __attribute__((ext_vector_type(8)))  _Float16 h8;
typedef __attribute__((ext_vector_type(4)))  _Float16 h4;
typedef __attribute__((ext_vector_type(8)))  float    v8f;

// ---------------- problem constants ----------------
#define CH   192      // H (channels)
#define NHD  2        // heads
#define HD   96       // head dim
#define NL   6        // layers
#define TT   2048     // T
#define FF   768      // ffn dim
#define IC   256      // in channels
#define NB   2        // batch
#define NOUT 192      // out channels (per half of stats)

__device__ __forceinline__ v8f zero8() {
  v8f z = {0.f, 0.f, 0.f, 0.f, 0.f, 0.f, 0.f, 0.f};
  return z;
}

// Load a 16-half WMMA operand as two contiguous 8-half (16B) LDS vectors.
// p0/p1 must be 16B aligned.
__device__ __forceinline__ v16h ldfrag(const _Float16* p0, const _Float16* p1) {
  h8 a = *(const h8*)p0;
  h8 b = *(const h8*)p1;
  return __builtin_shufflevector(a, b, 0, 1, 2, 3, 4, 5, 6, 7,
                                 8, 9, 10, 11, 12, 13, 14, 15);
}

__device__ __forceinline__ h4 cvt4(float4 f) {
  h4 h = {(_Float16)f.x, (_Float16)f.y, (_Float16)f.z, (_Float16)f.w};
  return h;
}

// =====================================================================
// Generic batched GEMM:  C[z, m, n] = act( sum_k A[m,k] * B[z,k,n] + bias[m] )
// A/B f32 in global, converted to f16 while staged into LDS; f32 accumulate
// via v_wmma_f32_16x16x32_f16.  B tile stored transposed [n][k] in LDS so
// each B-fragment is 16 contiguous halves per lane (2x ds_load_b128).
// B-fragment chain is software-pipelined (load nf+1 before WMMA nf) so the
// scheduler can use partial s_wait_dscnt instead of a full drain.
// bmode==1: implicit im2col for conv1d(K=3,pad=1): kk = c*3 + j,
//           B[z,kk,n] = (0<=n+j-1<Tc && n+j-1<len[z]) ? Bsrc[z, c, n+j-1] : 0
// flags: bit0 = ReLU, bit1 = zero columns n >= len[z]
// Block: 128 threads (4 waves), tile 64(M) x 64(N), K-step 32.
// =====================================================================
#define ASV 40   // A tile row stride (halves), mult of 8
#define BSV 40   // B^T tile row stride (halves), mult of 8
__global__ __launch_bounds__(128) void gemm_wmma_k(
    const float* __restrict__ A, long long lda_m, long long lda_k,
    const float* __restrict__ Bm, long long ldb_k, long long ldb_n, long long sB,
    const float* __restrict__ bias,
    float* __restrict__ C, long long ldc, long long sC,
    int Kd, int bmode, int Tc, const int* __restrict__ lengths, int flags)
{
  __shared__ __align__(16) _Float16 sa[64 * ASV];
  __shared__ __align__(16) _Float16 sbT[64 * BSV];

  const int tid = threadIdx.x;
  const int w   = tid >> 5;
  const int lane = tid & 31;
  const int hi  = lane >> 4;
  const int l15 = lane & 15;
  const int n0 = blockIdx.x * 64;
  const int m0 = blockIdx.y * 64;
  const int z  = blockIdx.z;
  const int lenz = lengths ? lengths[z] : 0x7fffffff;

  const float* Bz = Bm + (size_t)sB * z;
  const bool avec = (lda_k == 1);
  const bool bvec = (bmode == 0) && (ldb_n == 1);

  v8f acc[4];
#pragma unroll
  for (int nf = 0; nf < 4; ++nf) acc[nf] = zero8();

  for (int k0 = 0; k0 < Kd; k0 += 32) {
    __syncthreads();
    // ---- A tile 64x32 (row-major, stride ASV) ----
    if (avec) {
#pragma unroll
      for (int r = 0; r < 4; ++r) {
        int i = tid + r * 128;          // [0,512)
        int mm = i >> 3;                // 64 rows
        int q4 = i & 7;                 // 8 groups of 4 k
        float4 f = *(const float4*)&A[(long long)(m0 + mm) * lda_m + k0 + q4 * 4];
        *(h4*)&sa[mm * ASV + q4 * 4] = cvt4(f);
      }
    } else {
#pragma unroll
      for (int r = 0; r < 16; ++r) {
        int i = tid + r * 128;
        int kk = i & 31, mm = i >> 5;
        sa[mm * ASV + kk] =
            (_Float16)A[(long long)(m0 + mm) * lda_m + (long long)(k0 + kk) * lda_k];
      }
    }
    // ---- B tile 32x64 stored transposed: sbT[n][k] ----
    if (bvec) {
#pragma unroll
      for (int r = 0; r < 4; ++r) {
        int i = tid + r * 128;          // [0,512)
        int kk = i >> 4;                // 32 k rows
        int n4 = i & 15;                // 16 groups of 4 n
        float4 f = *(const float4*)&Bz[(long long)(k0 + kk) * ldb_k + n0 + n4 * 4];
        sbT[(n4 * 4 + 0) * BSV + kk] = (_Float16)f.x;
        sbT[(n4 * 4 + 1) * BSV + kk] = (_Float16)f.y;
        sbT[(n4 * 4 + 2) * BSV + kk] = (_Float16)f.z;
        sbT[(n4 * 4 + 3) * BSV + kk] = (_Float16)f.w;
      }
    } else {
#pragma unroll
      for (int r = 0; r < 16; ++r) {
        int i = tid + r * 128;
        int nn = i & 63, kk = i >> 6;
        int kg = k0 + kk, ng = n0 + nn;
        float v;
        if (bmode == 0) {
          v = Bz[(long long)kg * ldb_k + (long long)ng * ldb_n];
        } else {
          int c = kg / 3;
          int j = kg - 3 * c;
          int s = ng + j - 1;
          v = (s >= 0 && s < Tc && s < lenz) ? Bz[(long long)c * ldb_k + s] : 0.f;
        }
        sbT[nn * BSV + kk] = (_Float16)v;
      }
    }
    __syncthreads();

    const _Float16* arow = sa + (w * 16 + l15) * ASV;
    v16h af = ldfrag(arow + hi * 16, arow + hi * 16 + 8);
    const _Float16* bp0 = sbT + l15 * BSV + hi * 16;
    v16h cur = ldfrag(bp0, bp0 + 8);
#pragma unroll
    for (int nf = 0; nf < 4; ++nf) {
      v16h nxt = cur;
      if (nf < 3) {
        const _Float16* bp = sbT + ((nf + 1) * 16 + l15) * BSV + hi * 16;
        nxt = ldfrag(bp, bp + 8);
      }
      acc[nf] = __builtin_amdgcn_wmma_f32_16x16x32_f16(
          false, af, false, cur, (short)0, acc[nf], false, false);
      cur = nxt;
    }
  }

#pragma unroll
  for (int nf = 0; nf < 4; ++nf) {
    int ng = n0 + nf * 16 + l15;
#pragma unroll
    for (int vv = 0; vv < 8; ++vv) {
      int mg = m0 + w * 16 + vv + hi * 8;
      float val = acc[nf][vv];
      if (bias) val += bias[mg];
      if (flags & 1) val = fmaxf(val, 0.f);
      if (flags & 2) val = (ng < lenz) ? val : 0.f;
      C[(size_t)sC * z + (long long)mg * ldc + ng] = val;
    }
  }
}

// =====================================================================
// Flash attention with banded relative-position terms (|s-t| <= 10).
// grid = (T/64, NH, B), block = 128 (4 waves); each wave owns 16 query rows.
// q/k/v/out are [B, CH, T] f32 (head h occupies rows h*HD .. h*HD+95).
// LDS layouts chosen so every WMMA operand = 2x ds_load_b128:
//   lq [t][d] (A for S),  lk [s][d] (B for S),
//   lp [t][s] (A for PV), lv [d][s] (B for PV).
// Fragment chains software-pipelined as in the GEMM.
// =====================================================================
#define QSV 104
#define KSV 104
#define VSV 72
#define PSV 72
__global__ __launch_bounds__(128) void attn_k(
    const float* __restrict__ qg, const float* __restrict__ kg,
    const float* __restrict__ vg,
    const float* __restrict__ relk, const float* __restrict__ relv,
    const int* __restrict__ lengths, float* __restrict__ aout)
{
  __shared__ __align__(16) _Float16 lq[64 * QSV];
  __shared__ __align__(16) _Float16 lk[64 * KSV];
  __shared__ __align__(16) _Float16 lv[96 * VSV];
  __shared__ __align__(16) _Float16 lp[64 * PSV];
  __shared__ float rbias[64 * 21];
  __shared__ float pband[64 * 21];
  __shared__ float lm[64];
  __shared__ float ll[64];

  const int tid = threadIdx.x;
  const int w   = tid >> 5;
  const int lane = tid & 31;
  const int hi  = lane >> 4;
  const int l15 = lane & 15;
  const int bb = blockIdx.z, hh = blockIdx.y;
  const int qt0 = blockIdx.x * 64;
  const int len = lengths[bb];
  const float scale = 0.10206207261596575f;  // 1/sqrt(96)

  const float* qb = qg + ((size_t)bb * CH + hh * HD) * TT;
  const float* kb = kg + ((size_t)bb * CH + hh * HD) * TT;
  const float* vb = vg + ((size_t)bb * CH + hh * HD) * TT;
  float* ob = aout + ((size_t)bb * CH + hh * HD) * TT;

  // Q tile (pre-scaled), [t][d]; float4 along t, scatter b16 into LDS
#pragma unroll
  for (int r = 0; r < 12; ++r) {
    int i = tid + r * 128;              // [0,1536)
    int t4 = i & 15;                    // 16 groups of 4 t
    int d  = i >> 4;                    // 96
    float4 f = *(const float4*)&qb[(size_t)d * TT + qt0 + t4 * 4];
    lq[(t4 * 4 + 0) * QSV + d] = (_Float16)(f.x * scale);
    lq[(t4 * 4 + 1) * QSV + d] = (_Float16)(f.y * scale);
    lq[(t4 * 4 + 2) * QSV + d] = (_Float16)(f.z * scale);
    lq[(t4 * 4 + 3) * QSV + d] = (_Float16)(f.w * scale);
  }
  __syncthreads();
  // relative-K bias table: rbias[t][dd] = qs[t] . relk[dd]
  for (int i = tid; i < 64 * 21; i += 128) {
    int lt = i / 21, dd = i % 21;
    float s = 0.f;
    for (int d = 0; d < 96; ++d)
      s += (float)lq[lt * QSV + d] * relk[dd * 96 + d];
    rbias[i] = s;
  }

  float m_i[8], l_i[8];
  v8f acc[6];
#pragma unroll
  for (int vv = 0; vv < 8; ++vv) { m_i[vv] = -1e30f; l_i[vv] = 0.f; }
#pragma unroll
  for (int nf = 0; nf < 6; ++nf) acc[nf] = zero8();

  const int nkt = (len + 63) >> 6;
  for (int kt = 0; kt < nkt; ++kt) {
    const int s0 = kt * 64;
    __syncthreads();
    // K -> lk [s][d] (scatter), V -> lv [d][s] (contiguous h4 stores)
#pragma unroll
    for (int r = 0; r < 12; ++r) {
      int i = tid + r * 128;            // [0,1536)
      int s4 = i & 15;
      int d  = i >> 4;
      float4 kf = *(const float4*)&kb[(size_t)d * TT + s0 + s4 * 4];
      float4 vf = *(const float4*)&vb[(size_t)d * TT + s0 + s4 * 4];
      lk[(s4 * 4 + 0) * KSV + d] = (_Float16)kf.x;
      lk[(s4 * 4 + 1) * KSV + d] = (_Float16)kf.y;
      lk[(s4 * 4 + 2) * KSV + d] = (_Float16)kf.z;
      lk[(s4 * 4 + 3) * KSV + d] = (_Float16)kf.w;
      *(h4*)&lv[d * VSV + s4 * 4] = cvt4(vf);
    }
    __syncthreads();

    // S = Qs * K^T  (16x64 per wave)
    v8f S[4];
#pragma unroll
    for (int nf = 0; nf < 4; ++nf) S[nf] = zero8();
    const _Float16* qrow = lq + (w * 16 + l15) * QSV;
#pragma unroll
    for (int kk = 0; kk < 96; kk += 32) {
      v16h af = ldfrag(qrow + kk + hi * 16, qrow + kk + hi * 16 + 8);
      const _Float16* kp0 = lk + l15 * KSV + kk + hi * 16;
      v16h cur = ldfrag(kp0, kp0 + 8);
#pragma unroll
      for (int nf = 0; nf < 4; ++nf) {
        v16h nxt = cur;
        if (nf < 3) {
          const _Float16* kp = lk + ((nf + 1) * 16 + l15) * KSV + kk + hi * 16;
          nxt = ldfrag(kp, kp + 8);
        }
        S[nf] = __builtin_amdgcn_wmma_f32_16x16x32_f16(
            false, af, false, cur, (short)0, S[nf], false, false);
        cur = nxt;
      }
    }

    // bias + mask + online softmax, write P tile (f16)
#pragma unroll
    for (int vv = 0; vv < 8; ++vv) {
      int lrow = w * 16 + vv + hi * 8;
      int tg = qt0 + lrow;
      float sv[4];
#pragma unroll
      for (int nf = 0; nf < 4; ++nf) {
        int sg = s0 + nf * 16 + l15;
        float x = S[nf][vv];
        int dd = sg - tg;
        if (dd >= -10 && dd <= 10) x += rbias[lrow * 21 + dd + 10];
        if (sg >= len) x = -1e4f;
        sv[nf] = x;
      }
      float mx = fmaxf(fmaxf(sv[0], sv[1]), fmaxf(sv[2], sv[3]));
      for (int off = 1; off < 16; off <<= 1)
        mx = fmaxf(mx, __shfl_xor(mx, off, 32));
      float mnew = fmaxf(m_i[vv], mx);
      float al = __expf(m_i[vv] - mnew);
      float rs = 0.f;
#pragma unroll
      for (int nf = 0; nf < 4; ++nf) {
        float p = __expf(sv[nf] - mnew);
        sv[nf] = p;
        rs += p;
      }
      for (int off = 1; off < 16; off <<= 1)
        rs += __shfl_xor(rs, off, 32);
      l_i[vv] = l_i[vv] * al + rs;
      m_i[vv] = mnew;
#pragma unroll
      for (int nf = 0; nf < 6; ++nf) acc[nf][vv] *= al;
#pragma unroll
      for (int nf = 0; nf < 4; ++nf)
        lp[lrow * PSV + nf * 16 + l15] = (_Float16)sv[nf];
    }

    // O += P * V (wave-private P region; LDS ops in-order per wave)
    const _Float16* prow = lp + (w * 16 + l15) * PSV;
#pragma unroll
    for (int sc = 0; sc < 64; sc += 32) {
      v16h af = ldfrag(prow + sc + hi * 16, prow + sc + hi * 16 + 8);
      const _Float16* vp0 = lv + l15 * VSV + sc + hi * 16;
      v16h cur = ldfrag(vp0, vp0 + 8);
#pragma unroll
      for (int nf = 0; nf < 6; ++nf) {
        v16h nxt = cur;
        if (nf < 5) {
          const _Float16* vp = lv + ((nf + 1) * 16 + l15) * VSV + sc + hi * 16;
          nxt = ldfrag(vp, vp + 8);
        }
        acc[nf] = __builtin_amdgcn_wmma_f32_16x16x32_f16(
            false, af, false, cur, (short)0, acc[nf], false, false);
        cur = nxt;
      }
    }
  }

  // publish row stats for band recomputation
  if (l15 == 0) {
#pragma unroll
    for (int vv = 0; vv < 8; ++vv) {
      int lrow = w * 16 + vv + hi * 8;
      lm[lrow] = m_i[vv];
      ll[lrow] = l_i[vv];
    }
  }
  __syncthreads();

  // banded probabilities: p[t, t+dd-10] for dd=0..20
  if (tid < 64) {
    int tg = qt0 + tid;
    float mrow = lm[tid], lrow = ll[tid];
    for (int dd = 0; dd < 21; ++dd) {
      int sg = tg + dd - 10;
      float p = 0.f;
      if (sg >= 0 && sg < len) {
        float qk = 0.f;
        for (int d = 0; d < 96; ++d)
          qk += (float)lq[tid * QSV + d] * kb[(size_t)d * TT + sg];
        qk += rbias[tid * 21 + dd];
        p = __expf(qk - mrow) / lrow;
      }
      pband[tid * 21 + dd] = p;
    }
  }
  __syncthreads();

  // out[t, d] = O[t, d]/l + sum_dd pband[t][dd] * relv[dd][d]
#pragma unroll
  for (int nf = 0; nf < 6; ++nf) {
    int d = nf * 16 + l15;
    float rv[21];
#pragma unroll
    for (int dd = 0; dd < 21; ++dd) rv[dd] = relv[dd * 96 + d];
#pragma unroll
    for (int vv = 0; vv < 8; ++vv) {
      int lrow = w * 16 + vv + hi * 8;
      float o = acc[nf][vv] / l_i[vv];
      float add = 0.f;
#pragma unroll
      for (int dd = 0; dd < 21; ++dd) add += pband[lrow * 21 + dd] * rv[dd];
      ob[(size_t)d * TT + qt0 + lrow] = o + add;
    }
  }
}

// =====================================================================
// embedding epilogue: x = leaky( (emb_out + pitch_emb[pitch]) * sqrt(H) ) * mask
// =====================================================================
__global__ void embed_k(const float* __restrict__ xe,
                        const float* __restrict__ pemb,
                        const int* __restrict__ pitch,
                        const int* __restrict__ lengths,
                        float* __restrict__ x)
{
  int idx = blockIdx.x * blockDim.x + threadIdx.x;
  if (idx >= NB * CH * TT) return;
  int b = idx / (CH * TT);
  int r = idx % (CH * TT);
  int c = r / TT;
  int t = r % TT;
  float v = xe[idx] + pemb[pitch[b * TT + t] * CH + c];
  v *= 13.856406460551018f;            // sqrt(192)
  v = (v >= 0.f) ? v : 0.1f * v;       // LeakyReLU(0.1)
  if (t >= lengths[b]) v = 0.f;
  x[idx] = v;
}

// =====================================================================
// x = LayerNorm_channels(x + y) * g + b     (in-place on x, per (b,t) column)
// =====================================================================
__global__ void ln_res_k(float* __restrict__ x, const float* __restrict__ y,
                         const float* __restrict__ g, const float* __restrict__ bt)
{
  int idx = blockIdx.x * blockDim.x + threadIdx.x;
  if (idx >= NB * TT) return;
  int b = idx / TT, t = idx % TT;
  float* xp = x + (size_t)b * CH * TT + t;
  const float* yp = y + (size_t)b * CH * TT + t;
  float s = 0.f, s2 = 0.f;
  for (int c = 0; c < CH; ++c) {
    float v = xp[(size_t)c * TT] + yp[(size_t)c * TT];
    s += v;
    s2 += v * v;
  }
  float mean = s * (1.0f / CH);
  float var = fmaxf(s2 * (1.0f / CH) - mean * mean, 0.f);
  float inv = 1.0f / sqrtf(var + 1e-5f);
  for (int c = 0; c < CH; ++c) {
    float v = xp[(size_t)c * TT] + yp[(size_t)c * TT];
    xp[(size_t)c * TT] = (v - mean) * inv * g[c] + bt[c];
  }
}

// =====================================================================
// split stats [B,384,T] -> d_out = [m | logs | x_mask]
// =====================================================================
__global__ void split_k(const float* __restrict__ stats,
                        const int* __restrict__ lengths,
                        float* __restrict__ out)
{
  const int BOT = NB * NOUT * TT;
  int idx = blockIdx.x * blockDim.x + threadIdx.x;
  if (idx < BOT) {
    int b = idx / (NOUT * TT);
    int r = idx % (NOUT * TT);
    out[idx] = stats[(size_t)b * 2 * NOUT * TT + r];
    out[BOT + idx] = stats[(size_t)b * 2 * NOUT * TT + (size_t)NOUT * TT + r];
  }
  if (idx < NB * TT) {
    int b = idx / TT, t = idx % TT;
    out[2 * BOT + idx] = (t < lengths[b]) ? 1.0f : 0.0f;
  }
}

// =====================================================================
extern "C" void kernel_launch(void* const* d_in, const int* in_sizes, int n_in,
                              void* d_out, int out_size, void* d_ws, size_t ws_size,
                              hipStream_t stream) {
  (void)in_sizes; (void)n_in; (void)out_size; (void)ws_size;

  const float* phone    = (const float*)d_in[0];
  const float* emb_w    = (const float*)d_in[1];
  const float* emb_b    = (const float*)d_in[2];
  const float* pitch_em = (const float*)d_in[3];
  const float* qw = (const float*)d_in[4];
  const float* qbias = (const float*)d_in[5];
  const float* kw = (const float*)d_in[6];
  const float* kbias = (const float*)d_in[7];
  const float* vw = (const float*)d_in[8];
  const float* vbias = (const float*)d_in[9];
  const float* ow = (const float*)d_in[10];
  const float* obias = (const float*)d_in[11];
  const float* rel_k = (const float*)d_in[12];
  const float* rel_v = (const float*)d_in[13];
  const float* ln1_g = (const float*)d_in[14];
  const float* ln1_b = (const float*)d_in[15];
  const float* w1 = (const float*)d_in[16];
  const float* b1 = (const float*)d_in[17];
  const float* w2 = (const float*)d_in[18];
  const float* b2 = (const float*)d_in[19];
  const float* ln2_g = (const float*)d_in[20];
  const float* ln2_b = (const float*)d_in[21];
  const float* proj_w = (const float*)d_in[22];
  const float* proj_b = (const float*)d_in[23];
  const int* pitch = (const int*)d_in[24];
  const int* lengths = (const int*)d_in[25];

  const long long HT = (long long)CH * TT;   // 192*2048
  const long long FT = (long long)FF * TT;   // 768*2048

  float* ws = (float*)d_ws;
  float* x    = ws;                  // [B, CH, T]
  float* qbf  = x    + (size_t)NB * HT;
  float* kbf  = qbf  + (size_t)NB * HT;
  float* vbf  = kbf  + (size_t)NB * HT;
  float* ao   = vbf  + (size_t)NB * HT;
  float* ybf  = ao   + (size_t)NB * HT;
  float* hbf  = ybf  + (size_t)NB * HT;      // [B, FF, T]
  float* stat = hbf  + (size_t)NB * FT;      // [B, 384, T]

  dim3 blk(128);
  dim3 gP(TT / 64, CH / 64, NB);     // 32 x 3 x 2 for 192-row GEMMs

  // --- embedding: emb_out = emb_w @ phone^T + emb_b  (into qbf as temp) ---
  gemm_wmma_k<<<gP, blk, 0, stream>>>(
      emb_w, IC, 1,
      phone, 1, IC, (long long)TT * IC,
      emb_b, qbf, TT, HT, IC, 0, TT, nullptr, 0);
  embed_k<<<(NB * CH * TT + 255) / 256, 256, 0, stream>>>(
      qbf, pitch_em, pitch, lengths, x);

  for (int i = 0; i < NL; ++i) {
    const float* qw_i = qw + (size_t)i * CH * CH;
    const float* kw_i = kw + (size_t)i * CH * CH;
    const float* vw_i = vw + (size_t)i * CH * CH;
    const float* ow_i = ow + (size_t)i * CH * CH;

    gemm_wmma_k<<<gP, blk, 0, stream>>>(qw_i, CH, 1, x, TT, 1, HT,
                                        qbias + i * CH, qbf, TT, HT, CH, 0, TT, nullptr, 0);
    gemm_wmma_k<<<gP, blk, 0, stream>>>(kw_i, CH, 1, x, TT, 1, HT,
                                        kbias + i * CH, kbf, TT, HT, CH, 0, TT, nullptr, 0);
    gemm_wmma_k<<<gP, blk, 0, stream>>>(vw_i, CH, 1, x, TT, 1, HT,
                                        vbias + i * CH, vbf, TT, HT, CH, 0, TT, nullptr, 0);

    attn_k<<<dim3(TT / 64, NHD, NB), blk, 0, stream>>>(
        qbf, kbf, vbf, rel_k + (size_t)i * 21 * HD, rel_v + (size_t)i * 21 * HD,
        lengths, ao);

    gemm_wmma_k<<<gP, blk, 0, stream>>>(ow_i, CH, 1, ao, TT, 1, HT,
                                        obias + i * CH, ybf, TT, HT, CH, 0, TT, nullptr, 0);

    ln_res_k<<<(NB * TT + 255) / 256, 256, 0, stream>>>(
        x, ybf, ln1_g + i * CH, ln1_b + i * CH);

    // conv1: 192 -> 768, K=3, implicit im2col (Kd = 576), ReLU
    gemm_wmma_k<<<dim3(TT / 64, FF / 64, NB), blk, 0, stream>>>(
        w1 + (size_t)i * FF * CH * 3, CH * 3, 1,
        x, TT, 0, HT,
        b1 + i * FF, hbf, TT, FT, CH * 3, 1, TT, lengths, 1);

    // conv2: 768 -> 192, K=3, implicit im2col (Kd = 2304), output masked
    gemm_wmma_k<<<gP, blk, 0, stream>>>(
        w2 + (size_t)i * CH * FF * 3, FF * 3, 1,
        hbf, TT, 0, FT,
        b2 + i * CH, ybf, TT, HT, FF * 3, 1, TT, lengths, 2);

    ln_res_k<<<(NB * TT + 255) / 256, 256, 0, stream>>>(
        x, ybf, ln2_g + i * CH, ln2_b + i * CH);
  }

  // final projection 192 -> 384, masked columns
  gemm_wmma_k<<<dim3(TT / 64, (2 * NOUT) / 64, NB), blk, 0, stream>>>(
      proj_w, CH, 1, x, TT, 1, HT,
      proj_b, stat, TT, (long long)2 * NOUT * TT, CH, 0, TT, lengths, 2);

  split_k<<<(NB * NOUT * TT + 255) / 256, 256, 0, stream>>>(
      stat, lengths, (float*)d_out);
}
